// MatchLayer_14267881357504
// MI455X (gfx1250) — matrix operand order repaired and checked by reading, not compiled, not run on hardware
//
#include <hip/hip_runtime.h>
#include <hip/hip_bf16.h>

typedef __attribute__((ext_vector_type(16))) __bf16 v16bf;
typedef __attribute__((ext_vector_type(8)))  __bf16 v8bf;
typedef __attribute__((ext_vector_type(4)))  __bf16 v4bf;
typedef __attribute__((ext_vector_type(8)))  float  v8f;

#define NB   8
#define NL   2048
#define ND   128
#define NROW 16384          // NB*NL rows per tensor
#define EPSF 1e-6f

// ---------------------------------------------------------------------------
// Kernel 0: zero the normalized-text column-sum buffer S [2][NB][ND]
// ---------------------------------------------------------------------------
__global__ void zero_kernel(float* __restrict__ S) {
    int i = blockIdx.x * 256 + threadIdx.x;
    if (i < 2 * NB * ND) S[i] = 0.0f;
}

// ---------------------------------------------------------------------------
// Kernel 1: per-row fp32 normalization -> bf16 copy; text rows accumulate S;
// hypo rows store reciprocal norm. One wave (32 lanes) per row, 8 rows/block.
// blockIdx.y selects tensor: 0=text_fw 1=text_bw 2=hypo_fw 3=hypo_bw
// ---------------------------------------------------------------------------
__global__ __launch_bounds__(256) void prep_kernel(
    const float* __restrict__ t_fw, const float* __restrict__ t_bw,
    const float* __restrict__ h_fw, const float* __restrict__ h_bw,
    __bf16* __restrict__ nrm,      // [4][NROW][ND] bf16
    float*  __restrict__ S,        // [2][NB][ND]
    float*  __restrict__ hnorm)    // [2][NROW] reciprocal norms of hypo rows
{
    const int tens = blockIdx.y;
    const int wave = threadIdx.x >> 5;
    const int lane = threadIdx.x & 31;
    const int row  = blockIdx.x * 8 + wave;          // [0, NROW)

    const float* srcs[4] = {t_fw, t_bw, h_fw, h_bw};
    const float* src = srcs[tens] + (size_t)row * ND + lane * 4;
    float4 v = *(const float4*)src;

    float ss = v.x*v.x + v.y*v.y + v.z*v.z + v.w*v.w;
    #pragma unroll
    for (int m = 1; m < 32; m <<= 1) ss += __shfl_xor(ss, m);
    const float rn = rsqrtf(fmaxf(ss, EPSF));        // 1/sqrt(max(ss,EPS))

    const float nx = v.x*rn, ny = v.y*rn, nz = v.z*rn, nw = v.w*rn;
    v4bf o;
    o[0] = (__bf16)nx; o[1] = (__bf16)ny; o[2] = (__bf16)nz; o[3] = (__bf16)nw;
    *(v4bf*)(nrm + ((size_t)tens * NROW + row) * ND + lane * 4) = o;

    __shared__ float lds[8][ND];
    if (tens < 2) {                                  // text: accumulate S
        float* l = &lds[wave][lane * 4];
        l[0] = nx; l[1] = ny; l[2] = nz; l[3] = nw;
        __syncthreads();
        if (threadIdx.x < ND) {
            float s = 0.0f;
            #pragma unroll
            for (int w = 0; w < 8; ++w) s += lds[w][threadIdx.x];
            const int b = row >> 11;                 // same b for whole block
            atomicAdd(&S[(tens * NB + b) * ND + threadIdx.x], s);
        }
    } else {                                         // hypo: store 1/norm
        if (lane == 0) hnorm[(tens - 2) * NROW + row] = rn;
    }
}

// ---------------------------------------------------------------------------
// Kernel 2: fused cosine GEMM + max pooling via v_wmma_f32_16x16x32_bf16.
// grid = (16 h-tiles of 128 rows, NB, 2 dirs), 128 threads = 4 waves.
// Each wave: 32 hypo rows = TWO A register tiles, so every B fragment feeds
// two WMMAs (halved B bandwidth per FLOP, two independent acc chains hide
// the WMMA RAW hazard). 128 t-tiles of 16 cols, K=128 as 4 chained WMMAs.
// ---------------------------------------------------------------------------
__global__ __launch_bounds__(128) void maxgemm_kernel(
    const __bf16* __restrict__ nrm, float* __restrict__ out)
{
    const int dir   = blockIdx.z;
    const int b     = blockIdx.y;
    const int wave  = threadIdx.x >> 5;
    const int lane  = threadIdx.x & 31;
    const int lhalf = lane >> 4;                     // 0: lanes 0-15, 1: 16-31
    const int l16   = lane & 15;
    const int hbase = blockIdx.x * 128 + wave * 32;  // 32 rows per wave

    // A = hypo_norm (tensor 2+dir): tile0 rows hbase+l16, tile1 rows +16
    const __bf16* Ab0 = nrm + ((size_t)(2 + dir) * NROW + b * NL + hbase + l16) * ND;
    const __bf16* Ab1 = Ab0 + (size_t)16 * ND;
    // B = text_norm (tensor dir)
    const __bf16* Bb = nrm + ((size_t)dir * NROW + (size_t)b * NL) * ND;

    // A fragments: element j<8 -> K = 32c + 8*lhalf + j ; j>=8 -> K += 16
    v16bf a0[4], a1[4];
    #pragma unroll
    for (int c = 0; c < 4; ++c) {
        const int k0 = c * 32 + lhalf * 8;
        v8bf lo0 = *(const v8bf*)(Ab0 + k0);
        v8bf hi0 = *(const v8bf*)(Ab0 + k0 + 16);
        v8bf lo1 = *(const v8bf*)(Ab1 + k0);
        v8bf hi1 = *(const v8bf*)(Ab1 + k0 + 16);
        #pragma unroll
        for (int j = 0; j < 8; ++j) {
            a0[c][j] = lo0[j]; a0[c][8 + j] = hi0[j];
            a1[c][j] = lo1[j]; a1[c][8 + j] = hi1[j];
        }
    }

    float vmax0[8], vmax1[8];
    #pragma unroll
    for (int r = 0; r < 8; ++r) { vmax0[r] = -2.0f; vmax1[r] = -2.0f; }

    #pragma unroll 2
    for (int tb = 0; tb < NL; tb += 16) {
        // B fragment: col N = tb + l16, element j<8 -> K = 32c + 8*lhalf + j ; j>=8 -> K += 16
        const __bf16* Brow = Bb + (size_t)(tb + l16) * ND + lhalf * 8;
        v8f acc0 = {}, acc1 = {};
        #pragma unroll
        for (int c = 0; c < 4; ++c) {
            v8bf blo = *(const v8bf*)(Brow + c * 32);
            v8bf bhi = *(const v8bf*)(Brow + c * 32 + 16);
            v16bf bm;
            #pragma unroll
            for (int j = 0; j < 8; ++j) { bm[j] = blo[j]; bm[8 + j] = bhi[j]; }
            acc0 = __builtin_amdgcn_wmma_f32_16x16x32_bf16(
                false, a0[c], false, bm, (short)0, acc0, false, false);
            acc1 = __builtin_amdgcn_wmma_f32_16x16x32_bf16(
                false, a1[c], false, bm, (short)0, acc1, false, false);
        }
        #pragma unroll
        for (int r = 0; r < 8; ++r) {
            vmax0[r] = fmaxf(vmax0[r], acc0[r]);
            vmax1[r] = fmaxf(vmax1[r], acc1[r]);
        }
    }

    // reduce over the 16 columns held in each half-wave (xor stays in half)
    #pragma unroll
    for (int r = 0; r < 8; ++r) {
        #pragma unroll
        for (int m = 1; m < 16; m <<= 1) {
            vmax0[r] = fmaxf(vmax0[r], __shfl_xor(vmax0[r], m));
            vmax1[r] = fmaxf(vmax1[r], __shfl_xor(vmax1[r], m));
        }
    }

    // C layout: acc[r] is row M=r (lanes 0-15) / M=r+8 (lanes 16-31)
    if (l16 == 0) {
        float* o0 = out + dir * 32768 + b * NL + hbase + lhalf * 8;
        float* o1 = o0 + 16;
        #pragma unroll
        for (int r = 0; r < 8; ++r) { o0[r] = vmax0[r]; o1[r] = vmax1[r]; }
    }
}

// ---------------------------------------------------------------------------
// Kernel 3: mean pooling in exact fp32: mean[b,h] = dot(hypo_raw, S)*rn/2048.
// One wave per (dir, b, h) row.
// ---------------------------------------------------------------------------
__global__ __launch_bounds__(256) void mean_kernel(
    const float* __restrict__ h_fw, const float* __restrict__ h_bw,
    const float* __restrict__ S, const float* __restrict__ hnorm,
    float* __restrict__ out)
{
    const int w    = (blockIdx.x * 256 + threadIdx.x) >> 5;  // [0, 32768)
    const int lane = threadIdx.x & 31;
    const int dir  = w >> 14;
    const int rem  = w & (NROW - 1);
    const int b    = rem >> 11;

    const float* src = (dir ? h_bw : h_fw) + (size_t)rem * ND + lane * 4;
    const float* Sp  = S + (dir * NB + b) * ND + lane * 4;
    float4 x = *(const float4*)src;
    float4 s = *(const float4*)Sp;
    float d = x.x*s.x + x.y*s.y + x.z*s.z + x.w*s.w;
    #pragma unroll
    for (int m = 1; m < 32; m <<= 1) d += __shfl_xor(d, m);

    if (lane == 0)
        out[16384 + dir * 32768 + rem] = d * hnorm[w] * (1.0f / 2048.0f);
}

// ---------------------------------------------------------------------------
extern "C" void kernel_launch(void* const* d_in, const int* in_sizes, int n_in,
                              void* d_out, int out_size, void* d_ws, size_t ws_size,
                              hipStream_t stream) {
    const float* t_fw = (const float*)d_in[0];
    const float* t_bw = (const float*)d_in[1];
    const float* h_fw = (const float*)d_in[2];
    const float* h_bw = (const float*)d_in[3];
    float* out = (float*)d_out;

    // workspace layout
    __bf16* nrm   = (__bf16*)d_ws;                               // 16 MiB
    float*  S     = (float*)((char*)d_ws + (size_t)16777216);    // 8 KiB
    float*  hnorm = (float*)((char*)d_ws + (size_t)16777216 + 8192); // 128 KiB

    zero_kernel<<<8, 256, 0, stream>>>(S);
    prep_kernel<<<dim3(NROW / 8, 4), 256, 0, stream>>>(t_fw, t_bw, h_fw, h_bw,
                                                       nrm, S, hnorm);
    maxgemm_kernel<<<dim3(16, NB, 2), 128, 0, stream>>>(nrm, out);
    mean_kernel<<<(2 * NROW) / 8, 256, 0, stream>>>(h_fw, h_bw, S, hnorm, out);
}
